// SelfAttentionModule_65481071398161
// MI455X (gfx1250) — matrix operand order repaired and checked by reading, not compiled, not run on hardware
//
#include <hip/hip_runtime.h>
#include <hip/hip_bf16.h>

typedef __bf16 bf16;
typedef __attribute__((ext_vector_type(16))) __bf16 v16bf;
typedef __attribute__((ext_vector_type(8)))  float  v8f;
typedef __attribute__((ext_vector_type(4)))  int    v4i;

#define AS1 __attribute__((address_space(1)))
#define AS3 __attribute__((address_space(3)))

#if defined(__has_builtin)
#if __has_builtin(__builtin_amdgcn_global_load_async_to_lds_b128)
#define HAVE_ASYNC_LDS 1
#endif
#endif
#ifndef HAVE_ASYNC_LDS
#define HAVE_ASYNC_LDS 0
#endif

union FragU  { v16bf v; uint4 q[2]; };
union U16x8  { uint4 q; bf16 h[8]; };

template <typename T> struct IsBF16 { static constexpr bool v = false; };
template <> struct IsBF16<bf16>     { static constexpr bool v = true;  };

__device__ __forceinline__ bf16 f2bf(float x) { return (bf16)x; }

__device__ __forceinline__ v8f v8f_zero() {
  v8f z = {0.f, 0.f, 0.f, 0.f, 0.f, 0.f, 0.f, 0.f};
  return z;
}

// pack 4 floats -> 4 bf16, single 8-byte LDS store
__device__ __forceinline__ void store_bf16x4(bf16* dst, float x, float y,
                                             float z, float w) {
  union { bf16 h[4]; uint2 u; } p;
  p.h[0] = f2bf(x); p.h[1] = f2bf(y); p.h[2] = f2bf(z); p.h[3] = f2bf(w);
  *(uint2*)dst = p.u;
}

// 16-byte global->LDS copy; async (ASYNCcnt) when the toolchain exposes it.
__device__ __forceinline__ void async_copy16(const bf16* g, bf16* l) {
#if HAVE_ASYNC_LDS
  __builtin_amdgcn_global_load_async_to_lds_b128(
      (AS1 v4i*)(g), (AS3 v4i*)(l), 0, 0);
#else
  *(uint4*)l = *(const uint4*)g;
#endif
}

__device__ __forceinline__ void wait_async0() {
#if defined(__has_builtin) && __has_builtin(__builtin_amdgcn_s_wait_asynccnt)
  __builtin_amdgcn_s_wait_asynccnt(0);
#else
  asm volatile("s_wait_asynccnt 0" ::: "memory");
#endif
}

__device__ __forceinline__ v8f wmma_bf16(v16bf a, v16bf b, v8f c) {
  return __builtin_amdgcn_wmma_f32_16x16x32_bf16(
      false, a, false, b, (short)0, c, false, false);
}

// A-fragment (16x32 bf16) from LDS tile [row][k], padded stride.
__device__ __forceinline__ v16bf load_frag_a(const bf16* lds, int stride,
                                             int rowbase, int kbase) {
  const int lane = threadIdx.x & 31;
  const bf16* p = lds + (rowbase + (lane & 15)) * stride + kbase + ((lane >> 4) << 3);
  FragU f;
  f.q[0] = *(const uint4*)(p);
  f.q[1] = *(const uint4*)(p + 16);
  return f.v;
}

// B-fragment (32x16 bf16) from LDS tile stored transposed [n][k].
__device__ __forceinline__ v16bf load_frag_b(const bf16* lds, int stride,
                                             int nbase, int kbase) {
  const int lane = threadIdx.x & 31;
  const bf16* p = lds + (nbase + (lane & 15)) * stride + kbase + ((lane >> 4) << 3);
  FragU f;
  f.q[0] = *(const uint4*)(p);
  f.q[1] = *(const uint4*)(p + 16);
  return f.v;
}

// C = op(A) @ op(B). TA: A is [K][M]; TB: B is [N][K].
// bf16 WMMA compute; element types templated (f32 converts on stage,
// bf16 stages as straight copies -- async global->LDS when legal).
template <typename TAe, typename TBe, typename TCe, bool TA, bool TB>
__global__ void __launch_bounds__(256)
gemm_wmma(const TAe* __restrict__ A, const TBe* __restrict__ B,
          TCe* __restrict__ C, int M, int N, int K) {
  constexpr int BM = 128, BN = 128, BK = 64, LDT = 72;  // 144-byte padded rows
  constexpr bool ABF = IsBF16<TAe>::v;  // bf16 A, [M][K]: async copy path
  constexpr bool BBF = IsBF16<TBe>::v;  // bf16 B, [K][N]: transpose scatter
  static_assert(!(ABF && TA), "bf16 A path assumes no transpose");
  static_assert(!(BBF && TB), "bf16 B path assumes [K][N] layout");

  __shared__ bf16 As[2][BM * LDT];
  __shared__ bf16 Bs[2][BN * LDT];

  const int tid  = threadIdx.x;
  const int lane = tid & 31;
  const int wave = tid >> 5;
  const int wm   = wave & 3;   // 4 waves x 32 rows
  const int wn   = wave >> 2;  // 2 waves x 64 cols
  const int m0   = blockIdx.y * BM;
  const int n0   = blockIdx.x * BN;

  v8f acc[2][4];
#pragma unroll
  for (int i = 0; i < 2; ++i)
#pragma unroll
    for (int j = 0; j < 4; ++j) acc[i][j] = v8f_zero();

  float4 ra[8], rb[8];

  auto stageA_async = [&](int buf, int k0) {
    if constexpr (ABF) {
#pragma unroll
      for (int i = 0; i < 4; ++i) {
        const int f = tid + (i << 8);
        const int r = f >> 3, c8 = f & 7;  // 8 uint4 per 64-elem row
        async_copy16(A + (long)(m0 + r) * K + k0 + (c8 << 3),
                     &As[buf][r * LDT + (c8 << 3)]);
      }
    }
  };

  auto gloadA = [&](int k0) {
    if constexpr (!ABF) {
#pragma unroll
      for (int i = 0; i < 8; ++i) {
        const int f = tid + (i << 8);
        if (!TA) {  // A [M][K] f32
          const int r = f >> 4, c4 = f & 15;
          ra[i] = *(const float4*)(A + (long)(m0 + r) * K + k0 + (c4 << 2));
        } else {    // A [K][M] f32
          const int kr = f >> 5, cc = f & 31;
          ra[i] = *(const float4*)(A + (long)(k0 + kr) * M + m0 + (cc << 2));
        }
      }
    }
  };

  auto lstoreA = [&](int buf) {
    if constexpr (!ABF) {
#pragma unroll
      for (int i = 0; i < 8; ++i) {
        const int f = tid + (i << 8);
        if (!TA) {
          const int r = f >> 4, c4 = f & 15;
          store_bf16x4(&As[buf][r * LDT + (c4 << 2)], ra[i].x, ra[i].y, ra[i].z, ra[i].w);
        } else {
          const int kr = f >> 5, cc = f & 31;
          const float v4[4] = {ra[i].x, ra[i].y, ra[i].z, ra[i].w};
#pragma unroll
          for (int j = 0; j < 4; ++j) As[buf][((cc << 2) + j) * LDT + kr] = f2bf(v4[j]);
        }
      }
    }
  };

  uint4 rbq[4];
  auto gloadB = [&](int k0) {
    if constexpr (BBF) {  // B [K][N] bf16
#pragma unroll
      for (int i = 0; i < 4; ++i) {
        const int f = tid + (i << 8);
        const int kr = f >> 4, nc8 = f & 15;
        rbq[i] = *(const uint4*)(B + (long)(k0 + kr) * N + n0 + (nc8 << 3));
      }
    } else {
#pragma unroll
      for (int i = 0; i < 8; ++i) {
        const int f = tid + (i << 8);
        if (!TB) {  // B [K][N] f32
          const int kr = f >> 5, cc = f & 31;
          rb[i] = *(const float4*)(B + (long)(k0 + kr) * N + n0 + (cc << 2));
        } else {    // B [N][K] f32
          const int r = f >> 4, c4 = f & 15;
          rb[i] = *(const float4*)(B + (long)(n0 + r) * K + k0 + (c4 << 2));
        }
      }
    }
  };

  auto lstoreB = [&](int buf) {
    if constexpr (BBF) {  // scatter-transpose into [n][k]
#pragma unroll
      for (int i = 0; i < 4; ++i) {
        const int f = tid + (i << 8);
        const int kr = f >> 4, nc8 = f & 15;
        U16x8 u; u.q = rbq[i];
#pragma unroll
        for (int j = 0; j < 8; ++j) Bs[buf][((nc8 << 3) + j) * LDT + kr] = u.h[j];
      }
    } else {
#pragma unroll
      for (int i = 0; i < 8; ++i) {
        const int f = tid + (i << 8);
        if (!TB) {  // transpose-stage into [n][k]
          const int kr = f >> 5, cc = f & 31;
          const float v4[4] = {rb[i].x, rb[i].y, rb[i].z, rb[i].w};
#pragma unroll
          for (int j = 0; j < 4; ++j) Bs[buf][((cc << 2) + j) * LDT + kr] = f2bf(v4[j]);
        } else {    // already [n][k]
          const int r = f >> 4, c4 = f & 15;
          store_bf16x4(&Bs[buf][r * LDT + (c4 << 2)], rb[i].x, rb[i].y, rb[i].z, rb[i].w);
        }
      }
    }
  };

  auto compute = [&](int buf) {
#pragma unroll
    for (int ks = 0; ks < 2; ++ks) {  // BK=64 -> two 32-deep WMMA steps
      v16bf afr[2], bfr[4];
#pragma unroll
      for (int tm = 0; tm < 2; ++tm)
        afr[tm] = load_frag_a(As[buf], LDT, wm * 32 + tm * 16, ks * 32);
#pragma unroll
      for (int tn = 0; tn < 4; ++tn)
        bfr[tn] = load_frag_b(Bs[buf], LDT, wn * 64 + tn * 16, ks * 32);
#pragma unroll
      for (int tm = 0; tm < 2; ++tm)
#pragma unroll
        for (int tn = 0; tn < 4; ++tn)
          acc[tm][tn] = wmma_bf16(afr[tm], bfr[tn], acc[tm][tn]);
    }
  };

  const int nk = K / BK;
  gloadA(0);
  gloadB(0);
  stageA_async(0, 0);
  for (int kt = 0; kt < nk; ++kt) {
    lstoreA(kt & 1);
    lstoreB(kt & 1);
    if (ABF) wait_async0();  // own async tile portion resident before barrier
    __syncthreads();
    if (kt + 1 < nk) {
      gloadA((kt + 1) * BK);
      gloadB((kt + 1) * BK);
      stageA_async((kt + 1) & 1, (kt + 1) * BK);
    }
    compute(kt & 1);
  }

  const int half = lane >> 4, col = lane & 15;
#pragma unroll
  for (int tm = 0; tm < 2; ++tm)
#pragma unroll
    for (int tn = 0; tn < 4; ++tn)
#pragma unroll
      for (int g = 0; g < 8; ++g) {
        const int m = m0 + wm * 32 + tm * 16 + g + 8 * half;
        const int n = n0 + wn * 64 + tn * 16 + col;
        if constexpr (IsBF16<TCe>::v) C[(long)m * N + n] = f2bf(acc[tm][tn][g]);
        else                          C[(long)m * N + n] = acc[tm][tn][g];
      }
}

// Flash attention: one block = 64 query rows of one (b,h). 4 waves x 16 rows.
// Q/K/V/out all bf16; Q and K tiles staged with async global->LDS copies.
__global__ void __launch_bounds__(128)
attn_wmma(const bf16* __restrict__ Q, const bf16* __restrict__ Km,
          const bf16* __restrict__ V, const float* __restrict__ ent,
          bf16* __restrict__ O) {
  constexpr int L = 2048, D = 1024, HD = 64, LDQ = 72;
  __shared__ bf16 Qs[64 * LDQ];
  __shared__ bf16 Ks[64 * LDQ];  // [kv][hd]
  __shared__ bf16 Vs[64 * LDQ];  // transposed: [hd][kv]
  __shared__ bf16 Ps[64 * LDQ];  // probs, bf16

  const int tid  = threadIdx.x;
  const int lane = tid & 31;
  const int wave = tid >> 5;
  const int half = lane >> 4, col = lane & 15;
  const int q0 = blockIdx.x * 64;
  const int b  = blockIdx.y >> 4;
  const int h  = blockIdx.y & 15;
  const float scale = ent[h] * 0.125f;  // 1/sqrt(64)

  const long base = ((long)b * L) * D + (long)h * HD;

  // async-stage Q tile (64x64 bf16 = straight copy)
#pragma unroll
  for (int i = 0; i < 4; ++i) {
    const int f = tid + (i << 7);
    const int r = f >> 3, c8 = f & 7;
    async_copy16(Q + base + (long)(q0 + r) * D + (c8 << 3), &Qs[r * LDQ + (c8 << 3)]);
  }

  float mrow[8], lrow[8];
#pragma unroll
  for (int g = 0; g < 8; ++g) { mrow[g] = -1e30f; lrow[g] = 0.f; }
  v8f o[4];
#pragma unroll
  for (int tn = 0; tn < 4; ++tn) o[tn] = v8f_zero();

  for (int kv0 = 0; kv0 < L; kv0 += 64) {
    __syncthreads();  // previous tile fully consumed
    // K: async straight copy; V: transpose-scatter (overlaps the async)
#pragma unroll
    for (int i = 0; i < 4; ++i) {
      const int f = tid + (i << 7);
      const int r = f >> 3, c8 = f & 7;
      async_copy16(Km + base + (long)(kv0 + r) * D + (c8 << 3), &Ks[r * LDQ + (c8 << 3)]);
      U16x8 u;
      u.q = *(const uint4*)(V + base + (long)(kv0 + r) * D + (c8 << 3));
#pragma unroll
      for (int j = 0; j < 8; ++j) Vs[((c8 << 3) + j) * LDQ + r] = u.h[j];
    }
    wait_async0();
    __syncthreads();

    // S(16x64) = Q_wave(16x64) @ K^T : hd=64 -> 2 WMMA k-steps
    const v16bf qa0 = load_frag_a(Qs, LDQ, wave * 16, 0);
    const v16bf qa1 = load_frag_a(Qs, LDQ, wave * 16, 32);
    v8f s[4];
#pragma unroll
    for (int tn = 0; tn < 4; ++tn) {
      v8f z = v8f_zero();
      z = wmma_bf16(qa0, load_frag_b(Ks, LDQ, tn * 16, 0), z);
      z = wmma_bf16(qa1, load_frag_b(Ks, LDQ, tn * 16, 32), z);
      s[tn] = z * scale;
    }

    // online softmax (row stats live in 16-lane halves of the C layout)
#pragma unroll
    for (int g = 0; g < 8; ++g) {
      float mx = fmaxf(fmaxf(s[0][g], s[1][g]), fmaxf(s[2][g], s[3][g]));
#pragma unroll
      for (int msk = 1; msk < 16; msk <<= 1)
        mx = fmaxf(mx, __shfl_xor(mx, msk, 32));
      const float mnew = fmaxf(mrow[g], mx);
      const float corr = __expf(mrow[g] - mnew);
      float rs = 0.f;
#pragma unroll
      for (int tn = 0; tn < 4; ++tn) {
        const float p = __expf(s[tn][g] - mnew);
        s[tn][g] = p;
        rs += p;
      }
#pragma unroll
      for (int msk = 1; msk < 16; msk <<= 1)
        rs += __shfl_xor(rs, msk, 32);
      lrow[g] = lrow[g] * corr + rs;
      mrow[g] = mnew;
#pragma unroll
      for (int tn = 0; tn < 4; ++tn) o[tn][g] *= corr;
    }

    // spill P (bf16, per-wave region), reload as A-fragments
#pragma unroll
    for (int tn = 0; tn < 4; ++tn)
#pragma unroll
      for (int g = 0; g < 8; ++g)
        Ps[(wave * 16 + g + 8 * half) * LDQ + tn * 16 + col] = f2bf(s[tn][g]);
    asm volatile("s_wait_dscnt 0" ::: "memory");

    const v16bf pa0 = load_frag_a(Ps, LDQ, wave * 16, 0);
    const v16bf pa1 = load_frag_a(Ps, LDQ, wave * 16, 32);
#pragma unroll
    for (int tn = 0; tn < 4; ++tn) {
      o[tn] = wmma_bf16(pa0, load_frag_b(Vs, LDQ, tn * 16, 0), o[tn]);
      o[tn] = wmma_bf16(pa1, load_frag_b(Vs, LDQ, tn * 16, 32), o[tn]);
    }
  }

  // normalize, write bf16 [b][q][h*64 + n]
#pragma unroll
  for (int g = 0; g < 8; ++g) {
    const float inv = 1.f / lrow[g];
    const int m = q0 + wave * 16 + g + 8 * half;
#pragma unroll
    for (int tn = 0; tn < 4; ++tn)
      O[base + (long)m * D + tn * 16 + col] = f2bf(o[tn][g] * inv);
  }
}

extern "C" void kernel_launch(void* const* d_in, const int* in_sizes, int n_in,
                              void* d_out, int out_size, void* d_ws, size_t ws_size,
                              hipStream_t stream) {
  const float* X   = (const float*)d_in[0];
  const float* R   = (const float*)d_in[1];
  const float* ent = (const float*)d_in[2];
  const float* Wq  = (const float*)d_in[3];
  const float* Wk  = (const float*)d_in[4];
  const float* Wv  = (const float*)d_in[5];
  const float* Wo  = (const float*)d_in[6];
  float* out = (float*)d_out;

  const int D = 1024, Lr = 2048, Bb = 2;
  const int Mrows = Bb * Lr;  // 4096

  bf16* ws = (bf16*)d_ws;
  bf16* Af = ws;                     // 3 * D*D    folded weights (Wx^T @ R), bf16
  bf16* Qb = Af + 3L * D * D;        // Mrows*D bf16
  bf16* Kb = Qb + (long)Mrows * D;
  bf16* Vb = Kb + (long)Mrows * D;
  bf16* Oa = Vb + (long)Mrows * D;   // attention output, bf16

  dim3 blk(256);

  // Fold rotation into projection weights: Af_i = W_i^T @ R  (f32 in, bf16 out)
  dim3 gFold(D / 128, D / 128);
  gemm_wmma<float, float, bf16, true, false><<<gFold, blk, 0, stream>>>(Wq, R, Af + 0L * D * D, D, D, D);
  gemm_wmma<float, float, bf16, true, false><<<gFold, blk, 0, stream>>>(Wk, R, Af + 1L * D * D, D, D, D);
  gemm_wmma<float, float, bf16, true, false><<<gFold, blk, 0, stream>>>(Wv, R, Af + 2L * D * D, D, D, D);

  // Q/K/V = X @ Af_i   (f32 A, bf16 B, bf16 out)
  dim3 gProj(D / 128, Mrows / 128);
  gemm_wmma<float, bf16, bf16, false, false><<<gProj, blk, 0, stream>>>(X, Af + 0L * D * D, Qb, Mrows, D, D);
  gemm_wmma<float, bf16, bf16, false, false><<<gProj, blk, 0, stream>>>(X, Af + 1L * D * D, Kb, Mrows, D, D);
  gemm_wmma<float, bf16, bf16, false, false><<<gProj, blk, 0, stream>>>(X, Af + 2L * D * D, Vb, Mrows, D, D);

  // Flash attention per (b,h), 64 query rows per block
  attn_wmma<<<dim3(Lr / 64, Bb * 16), dim3(128), 0, stream>>>(Qb, Kb, Vb, ent, Oa);

  // Output projection: out = Oa @ Wo^T  (bf16 A -> async LDS staging, f32 out)
  gemm_wmma<bf16, float, float, false, true><<<gProj, blk, 0, stream>>>(Oa, Wo, out, Mrows, D, D);
}
